// CrossAttentionBridge_63582695850803
// MI455X (gfx1250) — matrix operand order repaired and checked by reading, not compiled, not run on hardware
//
#include <hip/hip_runtime.h>
#include <hip/hip_bf16.h>

typedef __attribute__((ext_vector_type(16))) _Float16 v16h;
typedef __attribute__((ext_vector_type(8)))  _Float16 v8h;
typedef __attribute__((ext_vector_type(8)))  float    v8f;

#define D_MODEL   1024
#define NUM_HEADS 16
#define HEAD_DIM  64
#define LQ        512
#define LKV       512
#define SCALE_F   0.125f     // HEAD_DIM^-0.5
#define LN_EPS_F  1e-5f

// concat two v8h halves into a v16h WMMA fragment (register-level)
__device__ __forceinline__ v16h cat16(v8h lo, v8h hi) {
    return __builtin_shufflevector(lo, hi, 0, 1, 2, 3, 4, 5, 6, 7,
                                   8, 9, 10, 11, 12, 13, 14, 15);
}

// ---------------- temporal bias (bilinear resize of analytic 128x128) -------
__device__ __forceinline__ float bias128(int i, int j) {
    float d = fabsf((float)(i - j));
    return __expf(-d * 0.1f) - d * 0.05f;
}
__device__ __forceinline__ float temporal_bias(int q, int k) {
    float x = (q + 0.5f) * 0.25f - 0.5f;   // 512 -> 128, half-pixel centers
    float y = (k + 0.5f) * 0.25f - 0.5f;
    x = fminf(fmaxf(x, 0.0f), 127.0f);
    y = fminf(fmaxf(y, 0.0f), 127.0f);
    int   x0 = (int)floorf(x), y0 = (int)floorf(y);
    int   x1 = min(x0 + 1, 127), y1 = min(y0 + 1, 127);
    float fx = x - (float)x0, fy = y - (float)y0;
    float v00 = bias128(x0, y0), v01 = bias128(x0, y1);
    float v10 = bias128(x1, y0), v11 = bias128(x1, y1);
    return v00 * (1.f - fx) * (1.f - fy) + v01 * (1.f - fx) * fy +
           v10 * fx * (1.f - fy) + v11 * fx * fy;
}

// precompute 512x512 bias table (1 MB, L2-resident, shared by all heads)
__global__ __launch_bounds__(256)
void bias_table_kernel(float* __restrict__ bt) {
    int t = blockIdx.x * 256 + threadIdx.x;
    int q = t >> 9, k = t & 511;
    bt[t] = temporal_bias(q, k);
}

// ---------------- LayerNorm + f16 cast (one block per row) ------------------
__global__ __launch_bounds__(256)
void ln_cast_kernel(const float* __restrict__ x, const float* __restrict__ gamma,
                    const float* __restrict__ beta, _Float16* __restrict__ out) {
    __shared__ float red[256];
    int row = blockIdx.x, tid = threadIdx.x;
    const float* xr = x + (size_t)row * D_MODEL;
    float v[4], s = 0.f;
#pragma unroll
    for (int i = 0; i < 4; ++i) { v[i] = xr[tid * 4 + i]; s += v[i]; }
    red[tid] = s; __syncthreads();
    for (int off = 128; off > 0; off >>= 1) {
        if (tid < off) red[tid] += red[tid + off];
        __syncthreads();
    }
    float mu = red[0] * (1.f / D_MODEL);
    __syncthreads();
    float sq = 0.f;
#pragma unroll
    for (int i = 0; i < 4; ++i) { float d = v[i] - mu; sq += d * d; }
    red[tid] = sq; __syncthreads();
    for (int off = 128; off > 0; off >>= 1) {
        if (tid < off) red[tid] += red[tid + off];
        __syncthreads();
    }
    float rstd = rsqrtf(red[0] * (1.f / D_MODEL) + LN_EPS_F);
    _Float16* orow = out + (size_t)row * D_MODEL;
#pragma unroll
    for (int i = 0; i < 4; ++i) {
        int c = tid * 4 + i;
        orow[c] = (_Float16)((v[i] - mu) * rstd * gamma[c] + beta[c]);
    }
}

// ---------------- W (NxK f32) -> f16 W^T in WMMA fragment tiles -------------
// Tile = 32(K) x 16(N); inner offset = (n&15)*32 + (k&31);
// tileIdx = (k>>5)*(N/16) + (n>>4).
__global__ __launch_bounds__(256)
void wtrans_kernel(const float* __restrict__ w, _Float16* __restrict__ wt) {
    int t = blockIdx.x * 256 + threadIdx.x;
    int n = t & (D_MODEL - 1);
    int k = t >> 10;
    size_t o = ((size_t)(k >> 5) * (D_MODEL / 16) + (n >> 4)) * 512 +
               (size_t)(n & 15) * 32 + (k & 31);
    wt[o] = (_Float16)w[(size_t)n * D_MODEL + k];
}

// ---------------- LDS-free f16 WMMA GEMM: C = A(MxK) * Bt + bias ------------
// Block = 8 waves; wave w computes rows [bm+w*16, +16) x cols [bn, bn+64).
__global__ __launch_bounds__(256)
void gemm_f16f32_kernel(const _Float16* __restrict__ A, const _Float16* __restrict__ Bt,
                        float* __restrict__ C, const float* __restrict__ bias,
                        int M, int N, int K) {
    int tid = threadIdx.x;
    int lane = tid & 31, w = tid >> 5;
    int half = lane >> 4, l15 = lane & 15;
    int bm = blockIdx.y * 128 + w * 16;
    int bn = blockIdx.x * 64;
    const int ntiles = N >> 4;

    v8f acc[4];
    v8f zero = {};
#pragma unroll
    for (int nt = 0; nt < 4; ++nt) acc[nt] = zero;

    const _Float16* arow = A + (size_t)(bm + l15) * K;
    for (int k0 = 0; k0 < K; k0 += 32) {
        v8h alo = *(const v8h*)&arow[k0 + half * 16];
        v8h ahi = *(const v8h*)&arow[k0 + half * 16 + 8];
        v16h af = cat16(alo, ahi);
        __builtin_prefetch(&arow[k0 + 32], 0, 1);
        const _Float16* btile = Bt + ((size_t)(k0 >> 5) * ntiles + (bn >> 4)) * 512 +
                                (size_t)l15 * 32 + half * 16;
#pragma unroll
        for (int nt = 0; nt < 4; ++nt) {
            v8h blo = *(const v8h*)&btile[(size_t)nt * 512];
            v8h bhi = *(const v8h*)&btile[(size_t)nt * 512 + 8];
            v16h bf = cat16(blo, bhi);
            acc[nt] = __builtin_amdgcn_wmma_f32_16x16x32_f16(false, af, false, bf,
                                                             (short)0, acc[nt], false, false);
        }
    }
    int gr = bm + half * 8;
#pragma unroll
    for (int nt = 0; nt < 4; ++nt) {
        int gc = bn + nt * 16 + l15;
        float bv = bias ? bias[gc] : 0.f;
#pragma unroll
        for (int r = 0; r < 8; ++r)
            C[(size_t)(gr + r) * N + gc] = acc[nt][r] + bv;
    }
}

// ---------------- RoPE + repack to per-head layouts -------------------------
// Q: [bh][l][d] row-major (A-side: contiguous K runs per row).
__global__ __launch_bounds__(256)
void rope_q_kernel(const float* __restrict__ qf, _Float16* __restrict__ q16) {
    int t = blockIdx.x * 256 + threadIdx.x;            // t = (bh*512 + l)*64 + d
    int d = t & 63, l = (t >> 6) & 511, bh = t >> 15;
    int b = bh >> 4, h = bh & 15, i = d & 31;
    const float* row = qf + (size_t)(b * 512 + l) * D_MODEL + h * 64;
    float xe = row[i], xo = row[i + 32];
    float ang = (float)l * __expf(-(float)i * 0.2878231366242557f);  // 1e4^(-i/32)
    float sn, cs; __sincosf(ang, &sn, &cs);
    q16[t] = (_Float16)((d < 32) ? (xe * cs - xo * sn) : (xe * sn + xo * cs));
}

// K^T in fragment tiles per (b,h): tile = 32(headdim) x 16(kv);
// tileIdx = (d>>5)*32 + (kv>>4); inner = (kv&15)*32 + (d&31).
__global__ __launch_bounds__(256)
void rope_kT_kernel(const float* __restrict__ kf, _Float16* __restrict__ kT) {
    int t = blockIdx.x * 256 + threadIdx.x;
    int kv = t & 511, d = (t >> 9) & 63, bh = t >> 15;
    int b = bh >> 4, h = bh & 15, i = d & 31;
    const float* row = kf + (size_t)(b * 512 + kv) * D_MODEL + h * 64;
    float xe = row[i], xo = row[i + 32];
    float ang = (float)kv * __expf(-(float)i * 0.2878231366242557f);
    float sn, cs; __sincosf(ang, &sn, &cs);
    float val = (d < 32) ? (xe * cs - xo * sn) : (xe * sn + xo * cs);
    size_t o = (size_t)bh * 32768 +
               ((size_t)(d >> 5) * 32 + (kv >> 4)) * 512 +
               (size_t)(kv & 15) * 32 + (d & 31);
    kT[o] = (_Float16)val;
}

// V in fragment tiles per (b,h): tile = 32(kv) x 16(headdim);
// tileIdx = (kv>>5)*4 + (d>>4); inner = (d&15)*32 + (kv&31).
__global__ __launch_bounds__(256)
void vcast_kernel(const float* __restrict__ vf, _Float16* __restrict__ v16) {
    int t = blockIdx.x * 256 + threadIdx.x;
    int d = t & 63, kv = (t >> 6) & 511, bh = t >> 15;
    int b = bh >> 4, h = bh & 15;
    float val = vf[(size_t)(b * 512 + kv) * D_MODEL + h * 64 + d];
    size_t o = (size_t)bh * 32768 +
               ((size_t)(kv >> 5) * 4 + (d >> 4)) * 512 +
               (size_t)(d & 15) * 32 + (kv & 31);
    v16[o] = (_Float16)val;
}

// ---------------- flash attention: 1 wave per (b,h,16-row q tile) -----------
__global__ __launch_bounds__(32)
void attn_kernel(const _Float16* __restrict__ q16, const _Float16* __restrict__ kT,
                 const _Float16* __restrict__ v16, const float* __restrict__ bt,
                 _Float16* __restrict__ att16) {
    __shared__ _Float16 pbuf[16][48];   // 96B row stride -> 16B-aligned b128 runs
    int lane = threadIdx.x;
    int half = lane >> 4, l15 = lane & 15;
    int bid = blockIdx.x;
    int qt = bid & 31, bh = bid >> 5;
    int b = bh >> 4, h = bh & 15;

    const _Float16* qp = q16 + ((size_t)bh * 512 + qt * 16) * 64;
    const _Float16* kbase = kT + (size_t)bh * 32768;
    const _Float16* vbase = v16 + (size_t)bh * 32768;

    v16h qa[2];
#pragma unroll
    for (int kh = 0; kh < 2; ++kh) {
        v8h lo = *(const v8h*)&qp[(size_t)l15 * 64 + kh * 32 + half * 16];
        v8h hi = *(const v8h*)&qp[(size_t)l15 * 64 + kh * 32 + half * 16 + 8];
        qa[kh] = cat16(lo, hi);
    }

    float rm[8], rl[8];
    v8f oacc[4];
    v8f zero = {};
#pragma unroll
    for (int r = 0; r < 8; ++r) { rm[r] = -1e30f; rl[r] = 0.f; }
#pragma unroll
    for (int n = 0; n < 4; ++n) oacc[n] = zero;

    // per-lane bias row pointers: row = qt*16 + half*8 + r, col = lane-local kc
    const float* btrow = bt + (size_t)(qt * 16 + half * 8) * 512 + l15;

    for (int c0 = 0; c0 < LKV; c0 += 32) {
        int ct = c0 >> 4;
        v8f s[2];
#pragma unroll
        for (int nt = 0; nt < 2; ++nt) {
            const _Float16* pk0 = kbase + ((size_t)(ct + nt)) * 512 + l15 * 32 + half * 16;
            const _Float16* pk1 = kbase + ((size_t)(32 + ct + nt)) * 512 + l15 * 32 + half * 16;
            v16h kb0 = cat16(*(const v8h*)pk0, *(const v8h*)(pk0 + 8));
            v16h kb1 = cat16(*(const v8h*)pk1, *(const v8h*)(pk1 + 8));
            v8f z = zero;
            z = __builtin_amdgcn_wmma_f32_16x16x32_f16(false, qa[0], false, kb0,
                                                       (short)0, z, false, false);
            z = __builtin_amdgcn_wmma_f32_16x16x32_f16(false, qa[1], false, kb1,
                                                       (short)0, z, false, false);
            s[nt] = z;
        }
        // scale + table-lookup bias (coalesced b32 loads from L2-resident table)
#pragma unroll
        for (int nt = 0; nt < 2; ++nt)
#pragma unroll
            for (int r = 0; r < 8; ++r)
                s[nt][r] = s[nt][r] * SCALE_F + btrow[(size_t)r * 512 + c0 + nt * 16];
        float alpha[8];
#pragma unroll
        for (int r = 0; r < 8; ++r) {
            float loc = fmaxf(s[0][r], s[1][r]);
#pragma unroll
            for (int m = 1; m < 16; m <<= 1) loc = fmaxf(loc, __shfl_xor(loc, m, 32));
            float mn = fmaxf(rm[r], loc);
            alpha[r] = __expf(rm[r] - mn);
            rm[r] = mn;
            float p0 = __expf(s[0][r] - mn);
            float p1 = __expf(s[1][r] - mn);
            s[0][r] = p0; s[1][r] = p1;
            float rs = p0 + p1;
#pragma unroll
            for (int m = 1; m < 16; m <<= 1) rs += __shfl_xor(rs, m, 32);
            rl[r] = rl[r] * alpha[r] + rs;
        }
#pragma unroll
        for (int n = 0; n < 4; ++n)
#pragma unroll
            for (int r = 0; r < 8; ++r) oacc[n][r] *= alpha[r];
        // repack P (C layout) -> A-fragment layout via LDS
#pragma unroll
        for (int nt = 0; nt < 2; ++nt)
#pragma unroll
            for (int r = 0; r < 8; ++r)
                pbuf[half * 8 + r][nt * 16 + l15] = (_Float16)s[nt][r];
        __syncthreads();
        v16h pa = cat16(*(const v8h*)&pbuf[l15][half * 16],
                        *(const v8h*)&pbuf[l15][half * 16 + 8]);
        __syncthreads();
#pragma unroll
        for (int n = 0; n < 4; ++n) {
            const _Float16* pv = vbase + ((size_t)((c0 >> 5) * 4 + n)) * 512 +
                                 l15 * 32 + half * 16;
            v16h vb = cat16(*(const v8h*)pv, *(const v8h*)(pv + 8));
            oacc[n] = __builtin_amdgcn_wmma_f32_16x16x32_f16(false, pa, false, vb,
                                                             (short)0, oacc[n], false, false);
        }
    }
#pragma unroll
    for (int n = 0; n < 4; ++n)
#pragma unroll
        for (int r = 0; r < 8; ++r) {
            int qr = qt * 16 + half * 8 + r;
            int d = n * 16 + l15;
            att16[(size_t)(b * 512 + qr) * D_MODEL + h * 64 + d] =
                (_Float16)(oacc[n][r] / rl[r]);
        }
}

// ---------------- sigmoid gate mix ------------------------------------------
__global__ __launch_bounds__(256)
void gate_mix_kernel(const float* __restrict__ proj, const float* __restrict__ gatel,
                     const float* __restrict__ resid, float* __restrict__ out, int n) {
    int t = blockIdx.x * 256 + threadIdx.x;
    if (t >= n) return;
    float a = proj[t];
    float g = 1.f / (1.f + __expf(-gatel[t]));
    out[t] = g * a + (1.f - g) * resid[t];
}

extern "C" void kernel_launch(void* const* d_in, const int* in_sizes, int n_in,
                              void* d_out, int out_size, void* d_ws, size_t ws_size,
                              hipStream_t stream) {
    (void)in_sizes; (void)n_in; (void)out_size; (void)ws_size;
    const float* dec   = (const float*)d_in[0];
    const float* enc   = (const float*)d_in[1];
    const float* Wqkv  = (const float*)d_in[2];
    const float* Wout  = (const float*)d_in[3];
    const float* bout  = (const float*)d_in[4];
    const float* Wgate = (const float*)d_in[5];
    const float* bgate = (const float*)d_in[6];
    const float* gamma = (const float*)d_in[7];
    const float* beta  = (const float*)d_in[8];
    float* out = (float*)d_out;

    const int M = 8 * LQ;                 // 4096 rows
    char* ws = (char*)d_ws;
    auto take = [&](size_t bytes) {
        char* p = ws;
        ws += (bytes + 255) & ~(size_t)255;
        return p;
    };
    _Float16* h16 = (_Float16*)take((size_t)M * D_MODEL * 2);
    _Float16* e16 = (_Float16*)take((size_t)M * D_MODEL * 2);
    _Float16* wq  = (_Float16*)take((size_t)D_MODEL * D_MODEL * 2);
    _Float16* wk  = (_Float16*)take((size_t)D_MODEL * D_MODEL * 2);
    _Float16* wv  = (_Float16*)take((size_t)D_MODEL * D_MODEL * 2);
    _Float16* wo  = (_Float16*)take((size_t)D_MODEL * D_MODEL * 2);
    _Float16* wg  = (_Float16*)take((size_t)D_MODEL * D_MODEL * 2);
    float*    bt  = (float*)take((size_t)LQ * LKV * 4);
    float*    qf  = (float*)take((size_t)M * D_MODEL * 4);
    float*    kf  = (float*)take((size_t)M * D_MODEL * 4);
    float*    vf  = (float*)take((size_t)M * D_MODEL * 4);
    _Float16* q16 = (_Float16*)take((size_t)M * D_MODEL * 2);
    _Float16* kTb = (_Float16*)take((size_t)M * D_MODEL * 2);
    _Float16* v16 = (_Float16*)take((size_t)M * D_MODEL * 2);
    _Float16* a16 = (_Float16*)take((size_t)M * D_MODEL * 2);
    float* proj  = qf;    // reuse: qf consumed before this is written
    float* gatel = kf;    // reuse

    dim3 blk256(256), blk32(32);
    ln_cast_kernel<<<M, blk256, 0, stream>>>(dec, gamma, beta, h16);
    ln_cast_kernel<<<M, blk256, 0, stream>>>(enc, gamma, beta, e16);
    bias_table_kernel<<<(LQ * LKV) / 256, blk256, 0, stream>>>(bt);
    wtrans_kernel<<<4096, blk256, 0, stream>>>(Wqkv, wq);
    wtrans_kernel<<<4096, blk256, 0, stream>>>(Wqkv + (size_t)D_MODEL * D_MODEL, wk);
    wtrans_kernel<<<4096, blk256, 0, stream>>>(Wqkv + (size_t)2 * D_MODEL * D_MODEL, wv);
    wtrans_kernel<<<4096, blk256, 0, stream>>>(Wout, wo);
    wtrans_kernel<<<4096, blk256, 0, stream>>>(Wgate, wg);

    dim3 gg(D_MODEL / 64, M / 128);
    gemm_f16f32_kernel<<<gg, blk256, 0, stream>>>(h16, wq, qf, nullptr, M, D_MODEL, D_MODEL);
    gemm_f16f32_kernel<<<gg, blk256, 0, stream>>>(e16, wk, kf, nullptr, M, D_MODEL, D_MODEL);
    gemm_f16f32_kernel<<<gg, blk256, 0, stream>>>(e16, wv, vf, nullptr, M, D_MODEL, D_MODEL);

    const int nelm = M * D_MODEL;
    rope_q_kernel<<<nelm / 256, blk256, 0, stream>>>(qf, q16);
    rope_kT_kernel<<<nelm / 256, blk256, 0, stream>>>(kf, kTb);
    vcast_kernel<<<nelm / 256, blk256, 0, stream>>>(vf, v16);

    attn_kernel<<<4096, blk32, 0, stream>>>(q16, kTb, v16, bt, a16);

    gemm_f16f32_kernel<<<gg, blk256, 0, stream>>>(a16, wo, proj, bout, M, D_MODEL, D_MODEL);
    gemm_f16f32_kernel<<<gg, blk256, 0, stream>>>(a16, wg, gatel, bgate, M, D_MODEL, D_MODEL);
    gate_mix_kernel<<<nelm / 256, blk256, 0, stream>>>(proj, gatel, dec, out, nelm);
}
